// SelectiveAttention_7876970021235
// MI455X (gfx1250) — matrix operand order repaired and checked by reading, not compiled, not run on hardware
//
#include <hip/hip_runtime.h>

// ---------------------------------------------------------------------------
// Selective attention for MI455X (gfx1250, wave32, WMMA).
// B=8, L=4096, D=1024, H=16, hd=64, ksel=409 (padded to 416 per batch).
// All GEMMs use v_wmma_f32_16x16x32_bf16 (fp32 accumulate).
// ---------------------------------------------------------------------------

#define BB    8
#define LL    4096
#define DD    1024
#define HH    16
#define HD    64
#define KSEL  409
#define MPAD  416          // 26 tiles of 16; batches never share a tile
#define D4    256
#define N3    3072

typedef __attribute__((ext_vector_type(16))) __bf16 v16bf;
typedef __attribute__((ext_vector_type(8)))  float  v8f;

union BF16x16 { uint4 q[2]; v16bf v; };

__device__ __forceinline__ v16bf load_b16x16(const __bf16* p) {
  BF16x16 t;
  t.q[0] = *(const uint4*)(p);
  t.q[1] = *(const uint4*)(p + 8);
  return t.v;
}

__device__ __forceinline__ v16bf cvt_f32x16(const float* p0, const float* p1) {
  float4 a0 = *(const float4*)(p0);
  float4 a1 = *(const float4*)(p0 + 4);
  float4 b0 = *(const float4*)(p1);
  float4 b1 = *(const float4*)(p1 + 4);
  v16bf v;
  v[0]=(__bf16)a0.x; v[1]=(__bf16)a0.y; v[2]=(__bf16)a0.z; v[3]=(__bf16)a0.w;
  v[4]=(__bf16)a1.x; v[5]=(__bf16)a1.y; v[6]=(__bf16)a1.z; v[7]=(__bf16)a1.w;
  v[8]=(__bf16)b0.x; v[9]=(__bf16)b0.y; v[10]=(__bf16)b0.z; v[11]=(__bf16)b0.w;
  v[12]=(__bf16)b1.x; v[13]=(__bf16)b1.y; v[14]=(__bf16)b1.z; v[15]=(__bf16)b1.w;
  return v;
}

#define WMMA_BF16(a, b, c) \
  __builtin_amdgcn_wmma_f32_16x16x32_bf16(false, (a), false, (b), (short)0, (c), false, false)

// ---------------------------------------------------------------------------
// Kernel 1: transpose + fp32->bf16 convert: src[K][N] row-major -> dst[N][K]
// ---------------------------------------------------------------------------
__global__ void sa_tconv_kernel(const float* __restrict__ src, __bf16* __restrict__ dst,
                                int K, int N) {
  int i = blockIdx.x * blockDim.x + threadIdx.x;
  if (i >= K * N) return;
  int n = i / K, k = i % K;
  dst[i] = (__bf16)src[(size_t)k * N + n];
}

// ---------------------------------------------------------------------------
// Kernel 2: importance scorer. One wave per 16-token tile.
// score = relu(x @ w1 + b1) . w2 + b2  (sigmoid dropped: monotonic for top-k)
// ---------------------------------------------------------------------------
__global__ __launch_bounds__(32) void sa_scorer_kernel(
    const float* __restrict__ x, const __bf16* __restrict__ w1T,
    const float* __restrict__ b1, const float* __restrict__ w2,
    const float* __restrict__ b2, float* __restrict__ scores) {
  const int tile = blockIdx.x;            // 0 .. B*L/16-1
  const int t0   = tile * 16;
  const int lane = threadIdx.x;
  const int mr   = lane & 15;
  const int hiL  = lane >> 4;

  float accs[8];
#pragma unroll
  for (int i = 0; i < 8; ++i) accs[i] = 0.f;

  const float* xrow = x + (size_t)(t0 + mr) * DD;

  for (int nt = 0; nt < D4 / 16; ++nt) {
    const int ncol = nt * 16 + mr;        // C-layout: this lane owns column ncol
    v8f c = {};
    for (int kc = 0; kc < DD; kc += 32) {
      v16bf a = cvt_f32x16(xrow + kc + hiL * 16, xrow + kc + hiL * 16 + 8);
      v16bf bm = load_b16x16(w1T + (size_t)ncol * DD + kc + hiL * 16);
      c = WMMA_BF16(a, bm, c);
    }
    const float b1v = b1[ncol];
    const float w2v = w2[ncol];
#pragma unroll
    for (int i = 0; i < 8; ++i) {
      float hv = c[i] + b1v;
      hv = hv > 0.f ? hv : 0.f;
      accs[i] += hv * w2v;
    }
  }

  __shared__ float red[32][8];
#pragma unroll
  for (int i = 0; i < 8; ++i) red[lane][i] = accs[i];
  __syncthreads();
  if (lane < 16) {
    float s = 0.f;
    if (lane < 8) {
      for (int l = 0; l < 16; ++l) s += red[l][lane];
    } else {
      for (int l = 16; l < 32; ++l) s += red[l][lane - 8];
    }
    scores[t0 + lane] = s + b2[0];
  }
}

// ---------------------------------------------------------------------------
// Kernel 3: per-batch top-k via 4-bit radix select on order-preserving uint keys
// ---------------------------------------------------------------------------
__device__ __forceinline__ unsigned sa_f2u(float f) {
  unsigned u = __float_as_uint(f);
  return (u & 0x80000000u) ? ~u : (u | 0x80000000u);
}

__global__ __launch_bounds__(256) void sa_topk_kernel(const float* __restrict__ scores,
                                                      int* __restrict__ top_idx) {
  const int b = blockIdx.x;
  const float* s = scores + (size_t)b * LL;
  const int t = threadIdx.x;
  __shared__ unsigned hist[16];
  __shared__ unsigned sprefix;
  __shared__ unsigned kept;
  __shared__ int pos_gt, pos_eq;

  if (t == 0) { sprefix = 0u; kept = KSEL; }
  __syncthreads();

  for (int shift = 28; shift >= 0; shift -= 4) {
    if (t < 16) hist[t] = 0u;
    __syncthreads();
    const unsigned pref = sprefix;
    const unsigned mask_hi = (shift == 28) ? 0u : (0xFFFFFFFFu << (shift + 4));
    for (int i = t; i < LL; i += 256) {
      unsigned u = sa_f2u(s[i]);
      if ((u & mask_hi) == pref) atomicAdd(&hist[(u >> shift) & 15], 1u);
    }
    __syncthreads();
    if (t == 0) {
      unsigned k = kept, cum = 0;
      int d = 15;
      for (; d >= 0; --d) {
        if (cum + hist[d] >= k) break;
        cum += hist[d];
      }
      if (d < 0) d = 0;
      kept = k - cum;
      sprefix = pref | ((unsigned)d << shift);
    }
    __syncthreads();
  }

  const unsigned thr = sprefix;
  const int need_eq = (int)kept;
  if (t == 0) { pos_gt = 0; pos_eq = 0; }
  __syncthreads();
  for (int i = t; i < LL; i += 256) {
    unsigned u = sa_f2u(s[i]);
    if (u > thr) {
      int p = atomicAdd(&pos_gt, 1);
      top_idx[b * KSEL + p] = i;
    } else if (u == thr) {
      int p = atomicAdd(&pos_eq, 1);
      if (p < need_eq) top_idx[b * KSEL + (KSEL - need_eq) + p] = i;
    }
  }
}

// ---------------------------------------------------------------------------
// Kernel 4: gather selected tokens -> bf16, zero-pad rows KSEL..MPAD-1
// ---------------------------------------------------------------------------
__global__ __launch_bounds__(256) void sa_gather_kernel(const float* __restrict__ x,
                                                        const int* __restrict__ top_idx,
                                                        __bf16* __restrict__ tokA) {
  const int row = blockIdx.x;       // 0 .. B*MPAD-1
  const int b = row / MPAD, j = row % MPAD;
  __bf16* d = tokA + (size_t)row * DD;
  if (j < KSEL) {
    const int src = top_idx[b * KSEL + j];
    const float* xs = x + ((size_t)b * LL + src) * DD;
    for (int c = threadIdx.x; c < DD; c += 256) d[c] = (__bf16)xs[c];
  } else {
    for (int c = threadIdx.x; c < DD; c += 256) d[c] = (__bf16)0.f;
  }
}

// ---------------------------------------------------------------------------
// Kernel 5: qkv GEMM.  [B*MPAD, 1024] x [1024, 3072] -> q,k [b][h][m][d] bf16,
// v transposed [b][h][d][m] bf16 (contiguous B-fragments for P@V).
// One wave computes a 16x64 strip.
// ---------------------------------------------------------------------------
__global__ __launch_bounds__(32) void sa_qkv_gemm_kernel(
    const __bf16* __restrict__ A, const __bf16* __restrict__ Bt,
    __bf16* __restrict__ q, __bf16* __restrict__ kmat, __bf16* __restrict__ vt) {
  const int mt = blockIdx.x;        // 0..207
  const int nc = blockIdx.y;        // 0..47
  const int lane = threadIdx.x;
  const int mr = lane & 15, hiL = lane >> 4;
  const int n0 = nc * 64;

  v8f c[4];
#pragma unroll
  for (int i = 0; i < 4; ++i) c[i] = {};

  const __bf16* arow = A + (size_t)(mt * 16 + mr) * DD;
  for (int kc = 0; kc < DD; kc += 32) {
    BF16x16 ta;
    ta.q[0] = *(const uint4*)(arow + kc + hiL * 16);
    ta.q[1] = *(const uint4*)(arow + kc + hiL * 16 + 8);
    const v16bf a = ta.v;
    const size_t ko = (size_t)kc + hiL * 16;
#pragma unroll
    for (int nt = 0; nt < 4; ++nt) {
      v16bf bm = load_b16x16(Bt + (size_t)(n0 + nt * 16 + mr) * DD + ko);
      c[nt] = WMMA_BF16(a, bm, c[nt]);
    }
  }

  const int gr0 = mt * 16;
#pragma unroll
  for (int nt = 0; nt < 4; ++nt) {
    const int col = n0 + nt * 16 + mr;
#pragma unroll
    for (int i = 0; i < 8; ++i) {
      const int gr = gr0 + i + 8 * hiL;
      const int b = gr / MPAD, m = gr % MPAD;
      const __bf16 val = (__bf16)c[nt][i];
      if (col < DD) {
        const int h = col >> 6, d = col & 63;
        q[(((size_t)b * HH + h) * MPAD + m) * HD + d] = val;
      } else if (col < 2 * DD) {
        const int c2 = col - DD, h = c2 >> 6, d = c2 & 63;
        kmat[(((size_t)b * HH + h) * MPAD + m) * HD + d] = val;
      } else {
        const int c3 = col - 2 * DD, h = c3 >> 6, d = c3 & 63;
        vt[(((size_t)b * HH + h) * HD + d) * MPAD + m] = val;
      }
    }
  }
}

// ---------------------------------------------------------------------------
// Kernel 6: attention, one wave per (b, h, 16-query tile).
// S = (Q K^T)/8 -> unnormalized softmax in LDS (bf16 P), out = (P V) * invsum.
// ---------------------------------------------------------------------------
__global__ __launch_bounds__(32) void sa_attn_kernel(
    const __bf16* __restrict__ q, const __bf16* __restrict__ kmat,
    const __bf16* __restrict__ vt, __bf16* __restrict__ attnout) {
  __shared__ __align__(16) float  S[16][MPAD];
  __shared__ __align__(16) __bf16 P[16][MPAD];
  __shared__ float pmax[16][2], psum[16][2], invs[16];

  const int qt = blockIdx.x, h = blockIdx.y, b = blockIdx.z;
  const int lane = threadIdx.x, mr = lane & 15, hiL = lane >> 4;
  const int m0 = qt * 16;

  const __bf16* qbh = q    + (((size_t)b * HH + h) * MPAD) * HD;
  const __bf16* kbh = kmat + (((size_t)b * HH + h) * MPAD) * HD;
  const __bf16* vbh = vt   + (((size_t)b * HH + h) * HD) * MPAD;

  v16bf a0, a1;
  {
    const __bf16* qr = qbh + (size_t)(m0 + mr) * HD;
    BF16x16 t;
    t.q[0] = *(const uint4*)(qr + hiL * 16);
    t.q[1] = *(const uint4*)(qr + hiL * 16 + 8);
    a0 = t.v;
    t.q[0] = *(const uint4*)(qr + 32 + hiL * 16);
    t.q[1] = *(const uint4*)(qr + 32 + hiL * 16 + 8);
    a1 = t.v;
  }

  for (int kt = 0; kt < MPAD / 16; ++kt) {
    const int n = kt * 16 + mr;                    // key row this lane owns
    const __bf16* kr = kbh + (size_t)n * HD;
    v16bf b0 = load_b16x16(kr + hiL * 16);         // d = 0..31
    v16bf b1 = load_b16x16(kr + 32 + hiL * 16);    // d = 32..63
    v8f c = {};
    c = WMMA_BF16(a0, b0, c);
    c = WMMA_BF16(a1, b1, c);
#pragma unroll
    for (int i = 0; i < 8; ++i) {
      const int m = i + 8 * hiL;
      S[m][n] = (n < KSEL) ? c[i] * 0.125f : -3.0e38f;
    }
  }
  __syncthreads();

  {   // row-max, two threads per row
    const int r = lane & 15, hh = lane >> 4;
    float mx = -3.4e38f;
    for (int c = hh * (MPAD / 2); c < (hh + 1) * (MPAD / 2); ++c)
      mx = fmaxf(mx, S[r][c]);
    pmax[r][hh] = mx;
  }
  __syncthreads();
  {   // exp + partial sum; keep P unnormalized, fold 1/sum into the output
    const int r = lane & 15, hh = lane >> 4;
    const float mxall = fmaxf(pmax[r][0], pmax[r][1]);
    float sm = 0.f;
    for (int c = hh * (MPAD / 2); c < (hh + 1) * (MPAD / 2); ++c) {
      const float e = __expf(S[r][c] - mxall);
      P[r][c] = (__bf16)e;
      sm += e;
    }
    psum[r][hh] = sm;
  }
  __syncthreads();
  if (lane < 16) invs[lane] = 1.f / (psum[lane][0] + psum[lane][1]);
  __syncthreads();

  v8f c[4];
#pragma unroll
  for (int i = 0; i < 4; ++i) c[i] = {};
  for (int kc = 0; kc < MPAD; kc += 32) {
    BF16x16 ta;
    ta.q[0] = *(const uint4*)&P[mr][kc + hiL * 16];
    ta.q[1] = *(const uint4*)&P[mr][kc + hiL * 16 + 8];
    const v16bf a = ta.v;
#pragma unroll
    for (int nd = 0; nd < 4; ++nd) {
      const int dcol = nd * 16 + mr;
      v16bf bm = load_b16x16(vbh + (size_t)dcol * MPAD + kc + hiL * 16);
      c[nd] = WMMA_BF16(a, bm, c[nd]);
    }
  }

#pragma unroll
  for (int nd = 0; nd < 4; ++nd)
#pragma unroll
    for (int i = 0; i < 8; ++i) {
      const int m = i + 8 * hiL;
      const size_t row = (size_t)b * MPAD + m0 + m;
      attnout[row * DD + h * HD + nd * 16 + mr] = (__bf16)(c[nd][i] * invs[m]);
    }
}

// ---------------------------------------------------------------------------
// Kernel 7: output projection GEMM: attnout[B*MPAD,1024] x wout -> proj fp32
// ---------------------------------------------------------------------------
__global__ __launch_bounds__(32) void sa_wout_gemm_kernel(
    const __bf16* __restrict__ A, const __bf16* __restrict__ Bt,
    float* __restrict__ proj) {
  const int mt = blockIdx.x;        // 0..207
  const int nc = blockIdx.y;        // 0..15
  const int lane = threadIdx.x;
  const int mr = lane & 15, hiL = lane >> 4;
  const int n0 = nc * 64;

  v8f c[4];
#pragma unroll
  for (int i = 0; i < 4; ++i) c[i] = {};

  const __bf16* arow = A + (size_t)(mt * 16 + mr) * DD;
  for (int kc = 0; kc < DD; kc += 32) {
    BF16x16 ta;
    ta.q[0] = *(const uint4*)(arow + kc + hiL * 16);
    ta.q[1] = *(const uint4*)(arow + kc + hiL * 16 + 8);
    const v16bf a = ta.v;
    const size_t ko = (size_t)kc + hiL * 16;
#pragma unroll
    for (int nt = 0; nt < 4; ++nt) {
      v16bf bm = load_b16x16(Bt + (size_t)(n0 + nt * 16 + mr) * DD + ko);
      c[nt] = WMMA_BF16(a, bm, c[nt]);
    }
  }

#pragma unroll
  for (int nt = 0; nt < 4; ++nt) {
    const int col = n0 + nt * 16 + mr;
#pragma unroll
    for (int i = 0; i < 8; ++i) {
      const int gr = mt * 16 + i + 8 * hiL;
      proj[(size_t)gr * DD + col] = c[nt][i];
    }
  }
}

// ---------------------------------------------------------------------------
// Kernel 8: out = x (vectorized copy)
// ---------------------------------------------------------------------------
__global__ __launch_bounds__(256) void sa_copy_kernel(const float4* __restrict__ x4,
                                                      float4* __restrict__ o4) {
  const size_t i = (size_t)blockIdx.x * blockDim.x + threadIdx.x;
  o4[i] = x4[i];
}

// ---------------------------------------------------------------------------
// Kernel 9: scatter residual: out[b, idx[j], :] += resw * proj[b, j, :]
// (indices unique -> no atomics)
// ---------------------------------------------------------------------------
__global__ __launch_bounds__(256) void sa_scatter_kernel(
    const float* __restrict__ proj, const int* __restrict__ top_idx,
    const float* __restrict__ resw, float* __restrict__ out) {
  const int r = blockIdx.x;         // 0 .. B*KSEL-1
  const int b = r / KSEL, j = r % KSEL;
  const int tok = top_idx[b * KSEL + j];
  const float* src = proj + ((size_t)b * MPAD + j) * DD;
  float* dst = out + ((size_t)b * LL + tok) * DD;
  const float rw = resw[0];
  for (int c = threadIdx.x; c < DD; c += 256) dst[c] += rw * src[c];
}

// ---------------------------------------------------------------------------
extern "C" void kernel_launch(void* const* d_in, const int* in_sizes, int n_in,
                              void* d_out, int out_size, void* d_ws, size_t ws_size,
                              hipStream_t stream) {
  (void)in_sizes; (void)n_in; (void)out_size; (void)ws_size;
  const float* x    = (const float*)d_in[0];
  const float* w1   = (const float*)d_in[1];
  const float* b1   = (const float*)d_in[2];
  const float* w2   = (const float*)d_in[3];
  const float* b2   = (const float*)d_in[4];
  const float* wqkv = (const float*)d_in[5];
  const float* wout = (const float*)d_in[6];
  const float* resw = (const float*)d_in[7];
  float* out = (float*)d_out;
  char* ws = (char*)d_ws;

  size_t off = 0;
  auto wsalloc = [&](size_t bytes) -> char* {
    char* p = ws + off;
    off += (bytes + 255) & ~(size_t)255;
    return p;
  };
  __bf16* w1T     = (__bf16*)wsalloc((size_t)D4 * DD * 2);
  __bf16* wqkvT   = (__bf16*)wsalloc((size_t)N3 * DD * 2);
  __bf16* woutT   = (__bf16*)wsalloc((size_t)DD * DD * 2);
  float*  scores  = (float*) wsalloc((size_t)BB * LL * 4);
  int*    topidx  = (int*)   wsalloc((size_t)BB * KSEL * 4);
  __bf16* tokA    = (__bf16*)wsalloc((size_t)BB * MPAD * DD * 2);
  __bf16* qbuf    = (__bf16*)wsalloc((size_t)BB * HH * MPAD * HD * 2);
  __bf16* kbuf    = (__bf16*)wsalloc((size_t)BB * HH * MPAD * HD * 2);
  __bf16* vtbuf   = (__bf16*)wsalloc((size_t)BB * HH * HD * MPAD * 2);
  __bf16* attnout = (__bf16*)wsalloc((size_t)BB * MPAD * DD * 2);
  float*  proj    = (float*) wsalloc((size_t)BB * MPAD * DD * 4);

  // 1. weight transpose + bf16 convert (all L2-resident afterwards)
  sa_tconv_kernel<<<(DD * D4 + 255) / 256, 256, 0, stream>>>(w1, w1T, DD, D4);
  sa_tconv_kernel<<<(DD * N3 + 255) / 256, 256, 0, stream>>>(wqkv, wqkvT, DD, N3);
  sa_tconv_kernel<<<(DD * DD + 255) / 256, 256, 0, stream>>>(wout, woutT, DD, DD);

  // 2. importance scores (WMMA bf16)
  sa_scorer_kernel<<<BB * LL / 16, 32, 0, stream>>>(x, w1T, b1, w2, b2, scores);

  // 3. per-batch top-k radix select
  sa_topk_kernel<<<BB, 256, 0, stream>>>(scores, topidx);

  // 4. gather + bf16 convert (zero-padded to MPAD rows/batch)
  sa_gather_kernel<<<BB * MPAD, 256, 0, stream>>>(x, topidx, tokA);

  // 5. qkv projection GEMM
  sa_qkv_gemm_kernel<<<dim3(BB * MPAD / 16, N3 / 64), 32, 0, stream>>>(
      tokA, wqkvT, qbuf, kbuf, vtbuf);

  // 6. attention over selected tokens
  sa_attn_kernel<<<dim3(MPAD / 16, HH, BB), 32, 0, stream>>>(qbuf, kbuf, vtbuf, attnout);

  // 7. output projection GEMM
  sa_wout_gemm_kernel<<<dim3(BB * MPAD / 16, DD / 64), 32, 0, stream>>>(
      attnout, woutT, proj);

  // 8. out = x
  sa_copy_kernel<<<(BB * LL * DD / 4) / 256, 256, 0, stream>>>(
      (const float4*)x, (float4*)out);

  // 9. residual scatter at selected positions
  sa_scatter_kernel<<<BB * KSEL, 256, 0, stream>>>(proj, topidx, resw, out);
}
